// MultiHeadLayer_33595234189450
// MI455X (gfx1250) — matrix operand order repaired and checked by reading, not compiled, not run on hardware
//
#include <hip/hip_runtime.h>

// ---------------------------------------------------------------------------
// MI455X (gfx1250) multi-head attention layer, bf16 WMMA pipeline with
// double-buffered LDS staging and async global->LDS copies (ASYNCcnt path).
// ---------------------------------------------------------------------------

typedef __bf16 bf16_t;
typedef __bf16  v16bf __attribute__((ext_vector_type(16)));
typedef __bf16  v8bf  __attribute__((ext_vector_type(8)));
typedef float   v8f   __attribute__((ext_vector_type(8)));
typedef float   v4f   __attribute__((ext_vector_type(4)));
typedef int     gv4i  __attribute__((vector_size(16)));   // matches builtin proto

#define AS1 __attribute__((address_space(1)))
#define AS3 __attribute__((address_space(3)))

#define B_   32
#define S_   256
#define H_   4096
#define NH_  16
#define D_   256
#define H3_  12288
#define NEG_INF_F (-3.402823466e38f)

#if defined(__has_builtin)
# if __has_builtin(__builtin_amdgcn_global_load_async_to_lds_b128)
#  define HAVE_ASYNC_LDS 1
# else
#  define HAVE_ASYNC_LDS 0
# endif
#else
# define HAVE_ASYNC_LDS 0
#endif

__device__ __forceinline__ bf16_t to_bf16(float f) {
    unsigned u = __builtin_bit_cast(unsigned, f);
    u += 0x7FFFu + ((u >> 16) & 1u);              // round-to-nearest-even
    return __builtin_bit_cast(bf16_t, (unsigned short)(u >> 16));
}

// 16-byte global -> LDS copy: async (ASYNCcnt, no VGPR data) when available.
__device__ __forceinline__ void cp16(const bf16_t* __restrict__ g,
                                     bf16_t* __restrict__ l) {
#if HAVE_ASYNC_LDS
    __builtin_amdgcn_global_load_async_to_lds_b128(
        (AS1 gv4i*)g, (AS3 gv4i*)l, 0, 0);
#else
    *(v8bf*)l = *(const v8bf*)g;
#endif
}

__device__ __forceinline__ void wait_async_copies() {
#if HAVE_ASYNC_LDS
# if __has_builtin(__builtin_amdgcn_s_wait_asynccnt)
    __builtin_amdgcn_s_wait_asynccnt(0);
# else
    asm volatile("s_wait_asynccnt 0x0" ::: "memory");
# endif
#endif
}

// ---------------------------------------------------------------------------
// Kernel 1: LayerNorm (f32 in) -> bf16 out.  One block (256 thr) per token row.
// If *isPre == 0, just converts to bf16 (post-LN handled at the end).
// ---------------------------------------------------------------------------
__global__ void __launch_bounds__(256)
k_ln_bf16(const float* __restrict__ x, const float* __restrict__ w,
          const float* __restrict__ b, const int* __restrict__ isPre,
          bf16_t* __restrict__ xn) {
    const int row = blockIdx.x;
    const float* xr = x + (size_t)row * H_;
    bf16_t* xo = xn + (size_t)row * H_;

    float v[16];
    float s = 0.f, s2 = 0.f;
    #pragma unroll
    for (int i = 0; i < 16; ++i) {
        float f = xr[threadIdx.x + i * 256];
        v[i] = f; s += f; s2 += f * f;
    }
    __shared__ float rs[256], rs2[256];
    rs[threadIdx.x] = s; rs2[threadIdx.x] = s2;
    __syncthreads();
    for (int off = 128; off > 0; off >>= 1) {
        if ((int)threadIdx.x < off) {
            rs[threadIdx.x]  += rs[threadIdx.x + off];
            rs2[threadIdx.x] += rs2[threadIdx.x + off];
        }
        __syncthreads();
    }
    const float mu   = rs[0] * (1.f / H_);
    const float var  = rs2[0] * (1.f / H_) - mu * mu;
    const float rstd = rsqrtf(var + 1e-5f);
    const bool  pre  = (*isPre != 0);

    #pragma unroll
    for (int i = 0; i < 16; ++i) {
        int c = threadIdx.x + i * 256;
        float f = pre ? ((v[i] - mu) * rstd * w[c] + b[c]) : v[i];
        xo[c] = to_bf16(f);
    }
}

// ---------------------------------------------------------------------------
// Shared 128x128 GEMM tile core.  256 threads = 8 wave32, wave tile 64x32
// (4 x 2 WMMA 16x16x32 bf16 tiles).  Double-buffered LDS: stage tile k+1
// (async global->LDS where possible) while WMMAs run on tile k.
//   A   : bf16 [M,K] row-major (lda)
//   Bsrc: TRANS_B ? [K,N] row-major (staged transposed; f32 converted on fly)
//                 : bf16 [N,K] row-major (straight async copy, e.g. K for QK^T)
//   LDS Bs is [n][k] so the B fragment (n = lane&15, K half*8..+7 and
//   16+half*8..+7) is two contiguous 8-wide loads, matching the ISA bf16 B layout.
// ---------------------------------------------------------------------------
template<bool OUT_BF16, bool DO_MASK, bool TRANS_B, typename TB>
__device__ __forceinline__ void
gemm128x128(const bf16_t* __restrict__ A, int lda,
            const TB* __restrict__ Bsrc, int ldb, int K,
            void* __restrict__ C, int ldc,
            float scale, const int* __restrict__ maskRow,
            int tile_m, int tile_n) {
    __shared__ __attribute__((aligned(64))) bf16_t As[2][128][32];
    __shared__ __attribute__((aligned(64))) bf16_t Bs[2][128][32];   // [n][k]

    const int tid  = threadIdx.x;
    const int lane = tid & 31;
    const int wid  = tid >> 5;
    const int wm   = wid & 1;       // 2 waves along M (2*64 = 128)
    const int wn   = wid >> 1;      // 4 waves along N (4*32 = 128)
    const int nrow = lane & 15;     // m (A) / n (B,C) index within 16
    const int half = lane >> 4;

    const bf16_t* Abase = A + (size_t)tile_m * 128 * lda;

    auto stage = [&](int buf, int k0) {
        // ---- A tile: 128 x 32 bf16, 2 x 16B chunks per thread (async) ------
        #pragma unroll
        for (int t = 0; t < 2; ++t) {
            int ci = tid * 2 + t;                 // 0..511
            int r  = ci >> 2;
            int k8 = (ci & 3) << 3;
            cp16(Abase + (size_t)r * lda + k0 + k8, &As[buf][r][k8]);
        }
        // ---- B tile into Bs[buf][n][k] -------------------------------------
        if constexpr (TRANS_B) {
            if constexpr (sizeof(TB) == 4) {      // f32 weights [K,N]: cvt+transpose
                #pragma unroll
                for (int t = 0; t < 4; ++t) {
                    int ci = tid + t * 256;       // 0..1023
                    int kk = ci >> 5;             // 0..31
                    int n4 = (ci & 31) << 2;      // 0..124
                    const float* gp = (const float*)Bsrc +
                        (size_t)(k0 + kk) * ldb + (size_t)tile_n * 128 + n4;
                    if (t == 0 && k0 + 32 < K)
                        __builtin_prefetch(gp + (size_t)32 * ldb, 0, 1);
                    v4f d = *(const v4f*)gp;
                    #pragma unroll
                    for (int e = 0; e < 4; ++e)
                        Bs[buf][n4 + e][kk] = to_bf16(d[e]);
                }
            } else {                              // bf16 [K,N] (V matrix): transpose
                #pragma unroll
                for (int t = 0; t < 2; ++t) {
                    int ci = tid * 2 + t;         // 0..511
                    int kk = ci >> 4;             // 0..31
                    int n8 = (ci & 15) << 3;      // 0..120
                    v8bf d = *(const v8bf*)((const bf16_t*)Bsrc +
                        (size_t)(k0 + kk) * ldb + (size_t)tile_n * 128 + n8);
                    #pragma unroll
                    for (int e = 0; e < 8; ++e) Bs[buf][n8 + e][kk] = d[e];
                }
            }
        } else {                                  // bf16 [N,K]: straight (async)
            #pragma unroll
            for (int t = 0; t < 2; ++t) {
                int ci = tid * 2 + t;
                int n  = ci >> 2;
                int k8 = (ci & 3) << 3;
                cp16((const bf16_t*)Bsrc +
                         (size_t)(tile_n * 128 + n) * ldb + k0 + k8,
                     &Bs[buf][n][k8]);
            }
        }
    };

    v8f acc[4][2] = {};
    const int nIter = K >> 5;

    stage(0, 0);
    wait_async_copies();
    __syncthreads();

    for (int it = 0; it < nIter; ++it) {
        const int cur = it & 1;
        if (it + 1 < nIter) stage(cur ^ 1, (it + 1) << 5);

        // ---- fragments per ISA VGPR layouts --------------------------------
        v16bf af[4], bfr[2];
        #pragma unroll
        for (int tm = 0; tm < 4; ++tm) {
            int r = wm * 64 + tm * 16 + nrow;
            v8bf lo = *(const v8bf*)&As[cur][r][half * 8];        // K half*8..+7
            v8bf hi = *(const v8bf*)&As[cur][r][16 + half * 8];   // K 16+half*8..
            af[tm] = __builtin_shufflevector(lo, hi,
                0,1,2,3,4,5,6,7,8,9,10,11,12,13,14,15);
        }
        #pragma unroll
        for (int tn = 0; tn < 2; ++tn) {
            int n = wn * 32 + tn * 16 + nrow;
            v8bf lo = *(const v8bf*)&Bs[cur][n][half * 8];
            v8bf hi = *(const v8bf*)&Bs[cur][n][16 + half * 8];
            bfr[tn] = __builtin_shufflevector(lo, hi,
                0,1,2,3,4,5,6,7,8,9,10,11,12,13,14,15);
        }
        // ---- 8 WMMA ops per wave per K-step --------------------------------
        #pragma unroll
        for (int tm = 0; tm < 4; ++tm)
            #pragma unroll
            for (int tn = 0; tn < 2; ++tn)
                acc[tm][tn] = __builtin_amdgcn_wmma_f32_16x16x32_bf16(
                    false, af[tm], false, bfr[tn],
                    (short)0, acc[tm][tn], false, false);

        wait_async_copies();       // next-tile async copies landed in LDS
        __syncthreads();           // + all waves done reading/writing this tile
    }

    // ---- epilogue: C layout is N = lane&15, M = v + 8*(lane>>4) ------------
    #pragma unroll
    for (int tm = 0; tm < 4; ++tm) {
        #pragma unroll
        for (int tn = 0; tn < 2; ++tn) {
            int ng = tile_n * 128 + wn * 32 + tn * 16 + nrow;
            bool masked = false;
            if constexpr (DO_MASK) masked = (maskRow[ng] != 0);
            #pragma unroll
            for (int v = 0; v < 8; ++v) {
                int mg = tile_m * 128 + wm * 64 + tm * 16 + half * 8 + v;
                float f = acc[tm][tn][v] * scale;
                if constexpr (DO_MASK) { if (masked) f = NEG_INF_F; }
                if constexpr (OUT_BF16)
                    ((bf16_t*)C)[(size_t)mg * ldc + ng] = to_bf16(f);
                else
                    ((float*)C)[(size_t)mg * ldc + ng] = f;
            }
        }
    }
}

// ---------------------------------------------------------------------------
// GEMM wrappers
// ---------------------------------------------------------------------------
// QKV: [8192,4096](bf16) x [4096,12288](f32) -> bf16 [8192,12288]
__global__ void __launch_bounds__(256)
k_gemm_qkv(const bf16_t* __restrict__ xn, const float* __restrict__ W,
           bf16_t* __restrict__ out) {
    gemm128x128<true, false, true, float>(xn, H_, W, H3_, H_,
                                          out, H3_, 1.f, nullptr,
                                          blockIdx.y, blockIdx.x);
}

// Scores per (b,h): Q[256,256] x K^T -> f32 [256,256], scaled + key mask.
__global__ void __launch_bounds__(256)
k_scores(const bf16_t* __restrict__ qkvo, const int* __restrict__ mask,
         float* __restrict__ scores) {
    const int z = blockIdx.z, b = z >> 4, h = z & 15;
    const bf16_t* Q  = qkvo + (size_t)b * S_ * H3_ + h * D_;
    const bf16_t* Km = Q + H_;                       // K block offset
    float* Cz = scores + (size_t)z * S_ * S_;
    gemm128x128<false, true, false, bf16_t>(Q, H3_, Km, H3_, D_,
                                            Cz, S_, 0.0625f /*1/sqrt(256)*/,
                                            mask + b * S_,
                                            blockIdx.y, blockIdx.x);
}

// ctx per (b,h): P[256,256](bf16) x V[256,256] -> bf16 into ctx[b, s, h*D..]
__global__ void __launch_bounds__(256)
k_ctx(const bf16_t* __restrict__ probs, const bf16_t* __restrict__ qkvo,
      bf16_t* __restrict__ ctx) {
    const int z = blockIdx.z, b = z >> 4, h = z & 15;
    const bf16_t* P = probs + (size_t)z * S_ * S_;
    const bf16_t* V = qkvo + (size_t)b * S_ * H3_ + 2 * H_ + h * D_;
    bf16_t* Cz = ctx + (size_t)b * S_ * H_ + h * D_;
    gemm128x128<true, false, true, bf16_t>(P, S_, V, H3_, S_,
                                           Cz, H_, 1.f, nullptr,
                                           blockIdx.y, blockIdx.x);
}

// O-proj: [8192,4096](bf16) x [4096,4096](f32) -> f32 d_out
__global__ void __launch_bounds__(256)
k_gemm_out(const bf16_t* __restrict__ ctx, const float* __restrict__ W,
           float* __restrict__ out) {
    gemm128x128<false, false, true, float>(ctx, H_, W, H_, H_,
                                           out, H_, 1.f, nullptr,
                                           blockIdx.y, blockIdx.x);
}

// ---------------------------------------------------------------------------
// Softmax: one wave32 per row of 256, f32 math, bf16 probs out.
// ---------------------------------------------------------------------------
__global__ void __launch_bounds__(256)
k_softmax(const float* __restrict__ scores, bf16_t* __restrict__ probs) {
    const int row  = blockIdx.x * 8 + (threadIdx.x >> 5);
    const int lane = threadIdx.x & 31;
    const float* sr = scores + (size_t)row * S_;

    float v[8];
    v4f d0 = *(const v4f*)(sr + lane * 8);
    v4f d1 = *(const v4f*)(sr + lane * 8 + 4);
    #pragma unroll
    for (int i = 0; i < 4; ++i) { v[i] = d0[i]; v[4 + i] = d1[i]; }

    float m = v[0];
    #pragma unroll
    for (int i = 1; i < 8; ++i) m = fmaxf(m, v[i]);
    #pragma unroll
    for (int off = 16; off > 0; off >>= 1) m = fmaxf(m, __shfl_xor(m, off, 32));

    float s = 0.f;
    #pragma unroll
    for (int i = 0; i < 8; ++i) { v[i] = __expf(v[i] - m); s += v[i]; }
    #pragma unroll
    for (int off = 16; off > 0; off >>= 1) s += __shfl_xor(s, off, 32);
    const float inv = 1.f / s;

    v8bf o;
    #pragma unroll
    for (int i = 0; i < 8; ++i) o[i] = to_bf16(v[i] * inv);
    *(v8bf*)(probs + (size_t)row * S_ + lane * 8) = o;
}

// ---------------------------------------------------------------------------
// Post-LN (only when *isPre == 0): in-place f32 LN per row of d_out.
// ---------------------------------------------------------------------------
__global__ void __launch_bounds__(256)
k_postln(float* __restrict__ out, const float* __restrict__ w,
         const float* __restrict__ b, const int* __restrict__ isPre) {
    if (*isPre != 0) return;                      // uniform branch
    const int row = blockIdx.x;
    float* xr = out + (size_t)row * H_;

    float v[16];
    float s = 0.f, s2 = 0.f;
    #pragma unroll
    for (int i = 0; i < 16; ++i) {
        float f = xr[threadIdx.x + i * 256];
        v[i] = f; s += f; s2 += f * f;
    }
    __shared__ float rs[256], rs2[256];
    rs[threadIdx.x] = s; rs2[threadIdx.x] = s2;
    __syncthreads();
    for (int off = 128; off > 0; off >>= 1) {
        if ((int)threadIdx.x < off) {
            rs[threadIdx.x]  += rs[threadIdx.x + off];
            rs2[threadIdx.x] += rs2[threadIdx.x + off];
        }
        __syncthreads();
    }
    const float mu   = rs[0] * (1.f / H_);
    const float var  = rs2[0] * (1.f / H_) - mu * mu;
    const float rstd = rsqrtf(var + 1e-5f);
    #pragma unroll
    for (int i = 0; i < 16; ++i) {
        int c = threadIdx.x + i * 256;
        xr[c] = (v[i] - mu) * rstd * w[c] + b[c];
    }
}

// ---------------------------------------------------------------------------
// Launch: LN -> QKV GEMM -> scores -> softmax -> PV -> O-proj -> (post-LN)
// Workspace (512 MiB): xn | qkv_out | scores | probs | ctx
// ---------------------------------------------------------------------------
extern "C" void kernel_launch(void* const* d_in, const int* in_sizes, int n_in,
                              void* d_out, int out_size, void* d_ws, size_t ws_size,
                              hipStream_t stream) {
    const float* inp    = (const float*)d_in[0];
    const int*   mask   = (const int*)  d_in[1];
    const float* weight = (const float*)d_in[2];
    const float* bias   = (const float*)d_in[3];
    const float* qkvW   = (const float*)d_in[4];
    const float* oW     = (const float*)d_in[5];
    const int*   isPre  = (const int*)  d_in[6];
    float* out = (float*)d_out;

    bf16_t* xn     = (bf16_t*)d_ws;                               // 64 MiB
    bf16_t* qkvo   = xn + (size_t)B_ * S_ * H_;                   // 192 MiB
    float*  scores = (float*)(qkvo + (size_t)B_ * S_ * H3_);      // 128 MiB
    bf16_t* probs  = (bf16_t*)(scores + (size_t)B_ * NH_ * S_ * S_); // 64 MiB
    bf16_t* ctx    = probs + (size_t)B_ * NH_ * S_ * S_;          // 64 MiB

    k_ln_bf16 <<<B_ * S_,            256, 0, stream>>>(inp, weight, bias, isPre, xn);
    k_gemm_qkv<<<dim3(H3_/128, (B_*S_)/128), 256, 0, stream>>>(xn, qkvW, qkvo);
    k_scores  <<<dim3(2, 2, B_ * NH_), 256, 0, stream>>>(qkvo, mask, scores);
    k_softmax <<<(B_ * NH_ * S_) / 8,  256, 0, stream>>>(scores, probs);
    k_ctx     <<<dim3(2, 2, B_ * NH_), 256, 0, stream>>>(probs, qkvo, ctx);
    k_gemm_out<<<dim3(H_/128, (B_*S_)/128), 256, 0, stream>>>(ctx, oW, out);
    k_postln  <<<B_ * S_,            256, 0, stream>>>(out, weight, bias, isPre);
}
